// Model_11458972746263
// MI455X (gfx1250) — matrix operand order repaired and checked by reading, not compiled, not run on hardware
//
#include <hip/hip_runtime.h>
#include <hip/hip_bf16.h>

// ---------------------------------------------------------------------------
// CDNA5 (gfx1250) implementation of the 2-stage Aligner reference.
// All large contractions run through v_wmma_f32_16x16x32_bf16 (wave32 WMMA),
// f32 accumulate, bf16 operands staged through LDS.
// Block tile 128x128, 8 waves, each wave owns 32x64 (2x4 WMMA 16x16 tiles).
// Out-of-range staging rows are CLAMPED (not guarded): clamped data only
// feeds outputs that the epilogue store-guards discard, so the k-loop has
// zero exec-mask manipulation.
// ---------------------------------------------------------------------------

typedef __attribute__((ext_vector_type(16))) __bf16 v16bf;
typedef __attribute__((ext_vector_type(8)))  float  v8f;

__device__ __forceinline__ unsigned short f2bf(float f) {
  unsigned int u = __float_as_uint(f);
  unsigned int r = (u + 0x7FFFu + ((u >> 16) & 1u)) >> 16;  // round-nearest-even
  return (unsigned short)r;
}

// ---------------------------------------------------------------------------
// Generic NT GEMM:  out[m,n] = act( alpha * sum_k A[m,k]*Bt[n,k] + bias[n] + Cin[m,n] )
// A  : bf16 row-major (M,K), lda = K
// Bt : bf16 row-major (N,K), ldb = K   (B stored transposed -> NT gemm)
// ---------------------------------------------------------------------------
#define BM 128
#define BN 128
#define BK 32
#define LDS_STRIDE 40   // 32 + 8 bf16 pad; row pitch = 80B (16B aligned)

__global__ __launch_bounds__(256) void gemm_nt_wmma(
    const unsigned short* __restrict__ A,  long long strideA,
    const unsigned short* __restrict__ Bt, long long strideB,
    const float* __restrict__ bias,
    const float* __restrict__ Cin, long long strideCin,
    const float* __restrict__ alphaPtr, int alphaIdx,
    float* __restrict__ outF, unsigned short* __restrict__ outB,
    long long strideOut,
    int M, int N, int K, int actRelu)
{
  __shared__ __align__(16) unsigned short As[BM * LDS_STRIDE];
  __shared__ __align__(16) unsigned short Bs[BN * LDS_STRIDE];

  const int z = blockIdx.z;
  const unsigned short* Ab = A  + (long long)z * strideA;
  const unsigned short* Bb = Bt + (long long)z * strideB;

  const int bm = blockIdx.y * BM;
  const int bn = blockIdx.x * BN;

  const int t    = threadIdx.x;
  const int wave = t >> 5;
  const int lane = t & 31;
  const int wm   = wave >> 1;   // 0..3 : 32 rows each
  const int wn   = wave & 1;    // 0..1 : 64 cols each
  const int lrow = lane & 15;
  const int lhi  = lane >> 4;

  // per-thread staging coordinates (2 chunks of 8 bf16 per tile per thread)
  const int row0 = t >> 2;          // 0..63
  const int row1 = row0 + 64;       // 64..127
  const int seg  = (t & 3) * 8;     // 0,8,16,24

  // clamp source rows into range; clamped lanes feed only discarded outputs
  const int ga0 = min(bm + row0, M - 1);
  const int ga1 = min(bm + row1, M - 1);
  const int gb0 = min(bn + row0, N - 1);
  const int gb1 = min(bn + row1, N - 1);
  const unsigned short* aptr0 = Ab + (long long)ga0 * K + seg;
  const unsigned short* aptr1 = Ab + (long long)ga1 * K + seg;
  const unsigned short* bptr0 = Bb + (long long)gb0 * K + seg;
  const unsigned short* bptr1 = Bb + (long long)gb1 * K + seg;

  v8f acc[2][4];
  for (int i = 0; i < 2; ++i)
    for (int j = 0; j < 4; ++j)
      for (int e = 0; e < 8; ++e) acc[i][j][e] = 0.0f;

  for (int k0 = 0; k0 < K; k0 += BK) {
    // --- stage A tile (128x32) and B tile (128x32), unconditional loads ---
    uint4 a0 = *(const uint4*)(aptr0 + k0);
    uint4 a1 = *(const uint4*)(aptr1 + k0);
    uint4 b0 = *(const uint4*)(bptr0 + k0);
    uint4 b1 = *(const uint4*)(bptr1 + k0);
    *(uint4*)(&As[row0 * LDS_STRIDE + seg]) = a0;
    *(uint4*)(&As[row1 * LDS_STRIDE + seg]) = a1;
    *(uint4*)(&Bs[row0 * LDS_STRIDE + seg]) = b0;
    *(uint4*)(&Bs[row1 * LDS_STRIDE + seg]) = b1;
    __syncthreads();

    // --- per-lane fragment loads per documented CDNA5 WMMA layouts ---
    union Frag { v16bf v; uint4 q[2]; };
    Frag afrag[2], bfrag[4];
#pragma unroll
    for (int i = 0; i < 2; ++i) {
      int row = wm * 32 + i * 16 + lrow;
      int kb  = lhi * 8;  // lanes 0-15: K=0..7,16..23 ; lanes 16-31: K=8..15,24..31
      afrag[i].q[0] = *(const uint4*)(&As[row * LDS_STRIDE + kb]);
      afrag[i].q[1] = *(const uint4*)(&As[row * LDS_STRIDE + kb + 16]);
    }
#pragma unroll
    for (int j = 0; j < 4; ++j) {
      int n  = wn * 64 + j * 16 + lrow;
      int kb = lhi * 8;  // lanes 0-15: K=0..7,16..23 ; lanes 16-31: K=8..15,24..31
      bfrag[j].q[0] = *(const uint4*)(&Bs[n * LDS_STRIDE + kb]);
      bfrag[j].q[1] = *(const uint4*)(&Bs[n * LDS_STRIDE + kb + 16]);
    }

#pragma unroll
    for (int i = 0; i < 2; ++i)
#pragma unroll
      for (int j = 0; j < 4; ++j)
        acc[i][j] = __builtin_amdgcn_wmma_f32_16x16x32_bf16(
            false, afrag[i].v, false, bfrag[j].v,
            (short)0, acc[i][j], false, false);

    __syncthreads();
  }

  const float alpha = alphaPtr ? alphaPtr[alphaIdx] : 1.0f;
  float*          oF = outF ? outF + (long long)z * strideOut : nullptr;
  unsigned short* oB = outB ? outB + (long long)z * strideOut : nullptr;
  const float*    Cb = Cin  ? Cin  + (long long)z * strideCin : nullptr;

  for (int i = 0; i < 2; ++i) {
    for (int j = 0; j < 4; ++j) {
      int gn = bn + wn * 64 + j * 16 + lrow;
      if (gn >= N) continue;
      for (int r = 0; r < 8; ++r) {
        int gm = bm + wm * 32 + i * 16 + lhi * 8 + r;
        if (gm >= M) continue;
        float v = alpha * acc[i][j][r];
        long long idx = (long long)gm * N + gn;
        if (bias) v += bias[gn];
        if (Cb)   v += Cb[idx];
        if (actRelu) v = v > 0.0f ? v : 0.0f;
        if (oF) oF[idx] = v;
        if (oB) oB[idx] = f2bf(v);
      }
    }
  }
}

// ---------------------------------------------------------------------------
// Support kernels
// ---------------------------------------------------------------------------

__global__ void cvt_f32_bf16_k(const float* __restrict__ in,
                               unsigned short* __restrict__ out, long long n) {
  long long i = (long long)blockIdx.x * blockDim.x + threadIdx.x;
  if (i < n) out[i] = f2bf(in[i]);
}

// in: f32 (batch,R,C) -> out: bf16 (batch,C,R)
__global__ void transpose_f32_bf16_k(const float* __restrict__ in,
                                     unsigned short* __restrict__ out,
                                     int R, int C) {
  __shared__ float tile[32][33];
  int z = blockIdx.z;
  const float* ib = in + (long long)z * R * C;
  unsigned short* ob = out + (long long)z * R * C;
  int c0 = blockIdx.x * 32, r0 = blockIdx.y * 32;
  int tx = threadIdx.x, ty = threadIdx.y;
  for (int i = ty; i < 32; i += 8) {
    int r = r0 + i, c = c0 + tx;
    tile[i][tx] = (r < R && c < C) ? ib[(long long)r * C + c] : 0.0f;
  }
  __syncthreads();
  for (int i = ty; i < 32; i += 8) {
    int c = c0 + i, r = r0 + tx;
    if (c < C && r < R) ob[(long long)c * R + r] = f2bf(tile[tx][i]);
  }
}

// Row softmax with column mask (mask==0 -> -1e9); out bf16 same layout.
// One wave (32 lanes) per row; X:(batch,Rb,C), colmask:(batch,C).
__global__ __launch_bounds__(256) void softmax_rows_k(
    const float* __restrict__ X, const int* __restrict__ colmask,
    unsigned short* __restrict__ out, int rowsPerBatch, int C, int totalRows) {
  int gid  = blockIdx.x * 8 + (threadIdx.x >> 5);
  int lane = threadIdx.x & 31;
  if (gid >= totalRows) return;
  int b = gid / rowsPerBatch;
  const float* x = X + (long long)gid * C;
  const int* mk  = colmask + (long long)b * C;
  unsigned short* o = out + (long long)gid * C;
  float vals[24];
  int nch = (C + 31) >> 5;
  float mx = -3.0e38f;
  for (int ch = 0; ch < nch; ++ch) {
    int j = ch * 32 + lane;
    float v = -3.0e38f;
    if (j < C) v = (mk[j] == 0) ? -1.0e9f : x[j];
    vals[ch] = v;
    mx = fmaxf(mx, v);
  }
  for (int off = 16; off > 0; off >>= 1) mx = fmaxf(mx, __shfl_xor(mx, off));
  float s = 0.0f;
  for (int ch = 0; ch < nch; ++ch) {
    int j = ch * 32 + lane;
    float e = (j < C) ? __expf(vals[ch] - mx) : 0.0f;
    vals[ch] = e;
    s += e;
  }
  for (int off = 16; off > 0; off >>= 1) s += __shfl_xor(s, off);
  float inv = 1.0f / s;
  for (int ch = 0; ch < nch; ++ch) {
    int j = ch * 32 + lane;
    if (j < C) o[j] = f2bf(vals[ch] * inv);
  }
}

// Column softmax (axis over R) with row mask; output TRANSPOSED bf16:
// X:(batch,R,C), rowmask:(batch,R)  ->  out:(batch,C,R), out[b][j][i]
__global__ __launch_bounds__(256) void softmax_cols_T_k(
    const float* __restrict__ X, const int* __restrict__ rowmask,
    unsigned short* __restrict__ out, int R, int C, int totalCols) {
  int gid  = blockIdx.x * 8 + (threadIdx.x >> 5);
  int lane = threadIdx.x & 31;
  if (gid >= totalCols) return;
  int b = gid / C;
  int j = gid - b * C;
  const float* x = X + (long long)b * R * C + j;
  const int* mk  = rowmask + (long long)b * R;
  unsigned short* o = out + (long long)b * C * R + (long long)j * R;
  float vals[24];
  int nch = (R + 31) >> 5;
  float mx = -3.0e38f;
  for (int ch = 0; ch < nch; ++ch) {
    int i = ch * 32 + lane;
    float v = -3.0e38f;
    if (i < R) v = (mk[i] == 0) ? -1.0e9f : x[(long long)i * C];
    vals[ch] = v;
    mx = fmaxf(mx, v);
  }
  for (int off = 16; off > 0; off >>= 1) mx = fmaxf(mx, __shfl_xor(mx, off));
  float s = 0.0f;
  for (int ch = 0; ch < nch; ++ch) {
    int i = ch * 32 + lane;
    float e = (i < R) ? __expf(vals[ch] - mx) : 0.0f;
    vals[ch] = e;
    s += e;
  }
  for (int off = 16; off > 0; off >>= 1) s += __shfl_xor(s, off);
  float inv = 1.0f / s;
  for (int ch = 0; ch < nch; ++ch) {
    int i = ch * 32 + lane;
    if (i < R) o[i] = f2bf(vals[ch] * inv);
  }
}

// m = [x, f, x*f, x-f] packed bf16 (rows, 4d)
__global__ void sfu_build_m_k(const float* __restrict__ x,
                              const float* __restrict__ f,
                              unsigned short* __restrict__ m,
                              long long rows, int d) {
  long long n = rows * (long long)d;
  long long i = (long long)blockIdx.x * blockDim.x + threadIdx.x;
  if (i >= n) return;
  long long r = i / d;
  int c = (int)(i - r * d);
  float xv = x[i], fv = f[i];
  unsigned short* mr = m + r * (long long)(4 * d);
  mr[c]         = f2bf(xv);
  mr[c + d]     = f2bf(fv);
  mr[c + 2 * d] = f2bf(xv * fv);
  mr[c + 3 * d] = f2bf(xv - fv);
}

// o = relu(r) * sigmoid(g) + (1 - sigmoid(g)) * x
__global__ void sfu_combine_k(const float* __restrict__ r,
                              const float* __restrict__ g,
                              const float* __restrict__ x,
                              float* __restrict__ outF,
                              unsigned short* __restrict__ outB, long long n) {
  long long i = (long long)blockIdx.x * blockDim.x + threadIdx.x;
  if (i >= n) return;
  float rv = fmaxf(r[i], 0.0f);
  float sg = 1.0f / (1.0f + __expf(-g[i]));
  float v  = rv * sg + (1.0f - sg) * x[i];
  if (outF) outF[i] = v;
  if (outB) outB[i] = f2bf(v);
}

__global__ void zero_diag_k(float* __restrict__ B, int c, int total) {
  int gid = blockIdx.x * blockDim.x + threadIdx.x;
  if (gid >= total) return;
  int b = gid / c, i = gid - b * c;
  B[(long long)b * c * c + (long long)i * c + i] = 0.0f;
}

// ---------------------------------------------------------------------------
// Host orchestration
// ---------------------------------------------------------------------------
static inline void launch_gemm(hipStream_t s,
    const unsigned short* A, long long sA,
    const unsigned short* Bt, long long sB,
    const float* bias, const float* Cin, long long sC,
    const float* alphaPtr, int alphaIdx,
    float* outF, unsigned short* outB, long long sO,
    int M, int N, int K, int relu, int batch) {
  dim3 grid((N + BN - 1) / BN, (M + BM - 1) / BM, batch);
  gemm_nt_wmma<<<grid, 256, 0, s>>>(A, sA, Bt, sB, bias, Cin, sC,
                                    alphaPtr, alphaIdx, outF, outB, sO,
                                    M, N, K, relu);
}

static inline void launch_transpose(hipStream_t s, const float* in,
                                    unsigned short* out, int R, int C, int batch) {
  dim3 grid((C + 31) / 32, (R + 31) / 32, batch);
  transpose_f32_bf16_k<<<grid, dim3(32, 8), 0, s>>>(in, out, R, C);
}

extern "C" void kernel_launch(void* const* d_in, const int* in_sizes, int n_in,
                              void* d_out, int out_size, void* d_ws, size_t ws_size,
                              hipStream_t stream) {
  (void)in_sizes; (void)n_in; (void)out_size; (void)ws_size;

  const int   Bn = 8, CL = 768, QL = 96, D = 768, D4 = 3072, T = 2;
  const long long BC = (long long)Bn * CL;        // 6144 rows
  const long long nx = BC * D;                    // 4,718,592
  const long long nv = (long long)Bn * QL * D;    //   589,824
  const long long nE = (long long)Bn * CL * QL;   //   589,824
  const long long nB = (long long)Bn * CL * CL;   // 4,718,592
  const long long nm = BC * D4;                   // 18,874,368

  const float* U  = (const float*)d_in[0];
  const float* V  = (const float*)d_in[1];
  const int* cmask = (const int*)d_in[2];
  const int* qmask = (const int*)d_in[3];
  const float* Wu  = (const float*)d_in[4];
  const float* bu  = (const float*)d_in[5];
  const float* Wv  = (const float*)d_in[6];
  const float* bv  = (const float*)d_in[7];
  const float* Wri = (const float*)d_in[8];
  const float* Wgi = (const float*)d_in[9];
  const float* gi  = (const float*)d_in[10];
  const float* W1  = (const float*)d_in[11];
  const float* b1  = (const float*)d_in[12];
  const float* W2  = (const float*)d_in[13];
  const float* b2  = (const float*)d_in[14];
  const float* Wrs = (const float*)d_in[15];
  const float* Wgs = (const float*)d_in[16];
  const float* gs  = (const float*)d_in[17];

  // ----- workspace carve -----
  char* w = (char*)d_ws;
  size_t off = 0;
  auto alloc = [&](size_t bytes) -> void* {
    void* p = w + off;
    off += (bytes + 255) & ~(size_t)255;
    return p;
  };
  // bf16 buffers
  unsigned short* xb    = (unsigned short*)alloc(nx * 2);           // x bf16
  unsigned short* Vb    = (unsigned short*)alloc(nv * 2);           // V bf16
  unsigned short* VbT   = (unsigned short*)alloc(nv * 2);           // V^T (b,d,q)
  unsigned short* Wsq   = (unsigned short*)alloc((size_t)D * D * 2);   // d x d weight^T
  unsigned short* Wbig  = (unsigned short*)alloc((size_t)D * D4 * 2);  // d x 4d weight^T
  unsigned short* Ctb   = (unsigned short*)alloc(nx * 2);
  unsigned short* Qtb   = (unsigned short*)alloc(nv * 2);
  unsigned short* EsTb  = (unsigned short*)alloc(nE * 2);           // (b,q,c)
  unsigned short* Bsb   = (unsigned short*)alloc(nB * 2);           // Bs / B2s
  unsigned short* Ettb  = (unsigned short*)alloc(nE * 2);
  unsigned short* mb    = (unsigned short*)alloc(nm * 2);
  unsigned short* hb16  = (unsigned short*)alloc(nx * 2);
  unsigned short* hbT   = (unsigned short*)alloc(nx * 2);           // (b,d,c)
  unsigned short* H1b   = (unsigned short*)alloc(nx * 2);
  unsigned short* H2b   = (unsigned short*)alloc(nx * 2);
  unsigned short* B1sTb = (unsigned short*)alloc(nB * 2);
  unsigned short* Bttb  = (unsigned short*)alloc(nB * 2);
  // f32 buffers
  float* Ebuf0 = (float*)alloc(nE * 4);
  float* Ebuf1 = (float*)alloc(nE * 4);
  float* Bbuf0 = (float*)alloc(nB * 4);
  float* Bbuf1 = (float*)alloc(nB * 4);
  float* qctx  = (float*)alloc(nx * 4);   // also hctx
  float* rbuf  = (float*)alloc(nx * 4);
  float* gbuf  = (float*)alloc(nx * 4);
  float* hf    = (float*)alloc(nx * 4);
  float* xZ    = (float*)alloc(nx * 4);

  const int THR = 256;
  auto gs1 = [&](long long n) { return (unsigned)((n + THR - 1) / THR); };

  // ----- one-time conversions -----
  cvt_f32_bf16_k<<<gs1(nx), THR, 0, stream>>>(U, xb, nx);
  cvt_f32_bf16_k<<<gs1(nv), THR, 0, stream>>>(V, Vb, nv);
  launch_transpose(stream, V, VbT, QL, D, Bn);   // (b,q,d) -> (b,d,q)

  float* Ebufs[2] = {Ebuf0, Ebuf1};
  float* Bbufs[2] = {Bbuf0, Bbuf1};
  const float* xf = U;     // current stage x (f32)
  float* prevE = nullptr;
  float* prevB = nullptr;

  for (int t = 0; t < T; ++t) {
    float* curE = Ebufs[t];
    float* curB = Bbufs[t];

    // ================= InterAlign =================
    // Ct = relu(x @ Wu + bu)  -> bf16
    launch_transpose(stream, Wu + (size_t)t * D * D, Wsq, D, D, 1);
    launch_gemm(stream, xb, 0, Wsq, 0, bu + (size_t)t * D, nullptr, 0,
                nullptr, 0, nullptr, Ctb, 0, (int)BC, D, D, 1, 1);
    // Qt = relu(V @ Wv + bv)  -> bf16
    launch_transpose(stream, Wv + (size_t)t * D * D, Wsq, D, D, 1);
    launch_gemm(stream, Vb, 0, Wsq, 0, bv + (size_t)t * D, nullptr, 0,
                nullptr, 0, nullptr, Qtb, 0, Bn * QL, D, D, 1, 1);
    // E0 = Ct @ Qt^T  (per batch) -> f32 curE
    launch_gemm(stream, Ctb, (long long)CL * D, Qtb, (long long)QL * D,
                nullptr, nullptr, 0, nullptr, 0, curE, nullptr,
                (long long)CL * QL, CL, QL, D, 0, Bn);
    if (prevE) {
      // Es^T (b,q,c), Bs (b,c,c); E = E0 + gi * Bs @ Es
      softmax_cols_T_k<<<(Bn * QL + 7) / 8, 256, 0, stream>>>(
          prevE, cmask, EsTb, CL, QL, Bn * QL);
      softmax_rows_k<<<(Bn * CL + 7) / 8, 256, 0, stream>>>(
          prevB, cmask, Bsb, CL, CL, Bn * CL);
      launch_gemm(stream, Bsb, (long long)CL * CL, EsTb, (long long)QL * CL,
                  nullptr, curE, (long long)CL * QL, gi, t,
                  curE, nullptr, (long long)CL * QL, CL, QL, CL, 0, Bn);
    }
    // Ett = softmax_q(E) -> bf16
    softmax_rows_k<<<(Bn * CL + 7) / 8, 256, 0, stream>>>(
        curE, qmask, Ettb, CL, QL, Bn * CL);
    // qctx = Ett @ V  (per batch) -> f32
    launch_gemm(stream, Ettb, (long long)CL * QL, VbT, (long long)D * QL,
                nullptr, nullptr, 0, nullptr, 0, qctx, nullptr,
                (long long)CL * D, CL, D, QL, 0, Bn);
    // SFU inter: h = relu(m@Wr)*sig(m@Wg) + (1-sig)*x
    sfu_build_m_k<<<gs1(nx), THR, 0, stream>>>(xf, qctx, mb, BC, D);
    launch_transpose(stream, Wri + (size_t)t * D4 * D, Wbig, D4, D, 1);
    launch_gemm(stream, mb, 0, Wbig, 0, nullptr, nullptr, 0, nullptr, 0,
                rbuf, nullptr, 0, (int)BC, D, D4, 1, 1);
    launch_transpose(stream, Wgi + (size_t)t * D4 * D, Wbig, D4, D, 1);
    launch_gemm(stream, mb, 0, Wbig, 0, nullptr, nullptr, 0, nullptr, 0,
                gbuf, nullptr, 0, (int)BC, D, D4, 0, 1);
    sfu_combine_k<<<gs1(nx), THR, 0, stream>>>(rbuf, gbuf, xf, hf, hb16, nx);

    // ================= SelfAlign =================
    launch_transpose(stream, W1 + (size_t)t * D * D, Wsq, D, D, 1);
    launch_gemm(stream, hb16, 0, Wsq, 0, b1 + (size_t)t * D, nullptr, 0,
                nullptr, 0, nullptr, H1b, 0, (int)BC, D, D, 1, 1);
    launch_transpose(stream, W2 + (size_t)t * D * D, Wsq, D, D, 1);
    launch_gemm(stream, hb16, 0, Wsq, 0, b2 + (size_t)t * D, nullptr, 0,
                nullptr, 0, nullptr, H2b, 0, (int)BC, D, D, 1, 1);
    // B0 = H1 @ H2^T -> f32 curB
    launch_gemm(stream, H1b, (long long)CL * D, H2b, (long long)CL * D,
                nullptr, nullptr, 0, nullptr, 0, curB, nullptr,
                (long long)CL * CL, CL, CL, D, 0, Bn);
    if (prevB) {
      softmax_cols_T_k<<<(Bn * CL + 7) / 8, 256, 0, stream>>>(
          prevB, cmask, B1sTb, CL, CL, Bn * CL);
      softmax_rows_k<<<(Bn * CL + 7) / 8, 256, 0, stream>>>(
          prevB, cmask, Bsb, CL, CL, Bn * CL);   // B2s
      launch_gemm(stream, Bsb, (long long)CL * CL, B1sTb, (long long)CL * CL,
                  nullptr, curB, (long long)CL * CL, gs, t,
                  curB, nullptr, (long long)CL * CL, CL, CL, CL, 0, Bn);
    }
    zero_diag_k<<<gs1(Bn * CL), THR, 0, stream>>>(curB, CL, Bn * CL);
    // Btt = softmax_c(B) -> bf16
    softmax_rows_k<<<(Bn * CL + 7) / 8, 256, 0, stream>>>(
        curB, cmask, Bttb, CL, CL, Bn * CL);
    // hctx = Btt @ h  (per batch); need h^T (b,d,c)
    launch_transpose(stream, hf, hbT, CL, D, Bn);
    launch_gemm(stream, Bttb, (long long)CL * CL, hbT, (long long)D * CL,
                nullptr, nullptr, 0, nullptr, 0, qctx /*hctx*/, nullptr,
                (long long)CL * D, CL, D, CL, 0, Bn);
    // SFU self -> Z
    sfu_build_m_k<<<gs1(nx), THR, 0, stream>>>(hf, qctx, mb, BC, D);
    launch_transpose(stream, Wrs + (size_t)t * D4 * D, Wbig, D4, D, 1);
    launch_gemm(stream, mb, 0, Wbig, 0, nullptr, nullptr, 0, nullptr, 0,
                rbuf, nullptr, 0, (int)BC, D, D4, 1, 1);
    launch_transpose(stream, Wgs + (size_t)t * D4 * D, Wbig, D4, D, 1);
    launch_gemm(stream, mb, 0, Wbig, 0, nullptr, nullptr, 0, nullptr, 0,
                gbuf, nullptr, 0, (int)BC, D, D4, 0, 1);
    float* zout = (t == T - 1) ? (float*)d_out : xZ;
    sfu_combine_k<<<gs1(nx), THR, 0, stream>>>(rbuf, gbuf, hf, zout, xb, nx);

    xf = xZ;
    prevE = curE;
    prevB = curB;
  }
}
